// WindowAttention_conv2qkv_84713934946804
// MI455X (gfx1250) — matrix operand order repaired and checked by reading, not compiled, not run on hardware
//
#include <hip/hip_runtime.h>
#include <hip/hip_bf16.h>

// ---------------------------------------------------------------------------
// WindowAttention fused pipeline for gfx1250 (MI455X), bf16 WMMA everywhere.
//   B=512, C=512, WS=16 -> N=256 tokens, NH=8 heads, HD=64, 3C=1536.
// ---------------------------------------------------------------------------

typedef __attribute__((ext_vector_type(16))) __bf16 v16bf;
typedef __attribute__((ext_vector_type(8)))  float  v8f;

union V16 { v16bf v; uint4 u[2]; };

#define BATCH   512
#define CDIM    512
#define NTOK    256
#define NHEAD   8
#define HDIM    64
#define TRIPLE  1536          // 3*C
#define SCALEQ  0.125f        // HD^-0.5

// ---- workspace layout (bytes) ---------------------------------------------
#define WB_BYTES   (TRIPLE * CDIM * 2)                       // 1.5 MB  bf16 W
#define XT_BYTES   ((size_t)BATCH * NTOK * CDIM * 2)         // 128 MB  bf16 x^T
#define BIAS_BYTES ((size_t)NHEAD * NTOK * NTOK * 4)         // 2 MB    f32 bias
#define Y_BYTES    ((size_t)BATCH * TRIPLE * NTOK * 2)       // 384 MB  bf16 qkv

// ---- attention LDS layout --------------------------------------------------
#define VSTR  264                         // 256 + 8 pad (bank rotate)
#define PSTR  40                          // 32 + 8 pad
#define OSTR  68                          // 64 + 4 pad
#define VT_BYTES   (64 * VSTR * 2)        // 33792
#define PST_BYTES  (16 * 16 * PSTR * 2)   // 20480 (16 waves x 16 rows)
#define OST_BYTES  (NTOK * OSTR * 4)      // 69632
#define ATTN_SMEM  (VT_BYTES + PST_BYTES + OST_BYTES)   // 123904 < 320KB WGP LDS

// A-fragment (16x32 bf16): lane r = row, runs of 8 at k0+8*half and k0+16+8*half
__device__ __forceinline__ v16bf ld_afrag(const __bf16* rowp_k0, int hf) {
    const __bf16* p = rowp_k0 + 8 * hf;
    V16 t;
    t.u[0] = *(const uint4*)(p);
    t.u[1] = *(const uint4*)(p + 16);
    return t.v;
}
// B-fragment (32x16 bf16): lane r = column, runs of 8 at k0+8*half and k0+16+8*half
__device__ __forceinline__ v16bf ld_bfrag(const __bf16* colp_k0, int hf) {
    const __bf16* p = colp_k0 + 8 * hf;
    V16 t;
    t.u[0] = *(const uint4*)(p);
    t.u[1] = *(const uint4*)(p + 16);
    return t.v;
}

// ---------------------------------------------------------------------------
// Kernel 0: W (1536x512 f32) -> bf16, folding softmax scale into q rows.
// q channels are o with (o % 6) in {0,1} under the torch reshape.
// ---------------------------------------------------------------------------
__global__ __launch_bounds__(256) void wa_cvt_w(const float* __restrict__ W,
                                                __bf16* __restrict__ Wb) {
    int i = blockIdx.x * 256 + threadIdx.x;          // < 1536*512
    int o = i >> 9;
    float s = ((o % 6) < 2) ? SCALEQ : 1.0f;
    Wb[i] = (__bf16)(W[i] * s);
}

// ---------------------------------------------------------------------------
// Kernel 1: x (B,512,256) f32 -> x^T (B,256,512) bf16, LDS tile transpose.
// Makes the GEMM B-fragments contiguous along K.
// ---------------------------------------------------------------------------
__global__ __launch_bounds__(256) void wa_cvt_xT(const float* __restrict__ x,
                                                 __bf16* __restrict__ xT) {
    __shared__ float t[32][33];
    int b = blockIdx.z, c0 = blockIdx.y * 32, n0 = blockIdx.x * 32;
    int tx = threadIdx.x, ty = threadIdx.y;          // 32 x 8
    const float* xb = x + ((size_t)b * CDIM + c0) * NTOK + n0;
#pragma unroll
    for (int yy = 0; yy < 32; yy += 8)
        t[ty + yy][tx] = xb[(size_t)(ty + yy) * NTOK + tx];
    __syncthreads();
    __bf16* xo = xT + ((size_t)b * NTOK + n0) * CDIM + c0;
#pragma unroll
    for (int yy = 0; yy < 32; yy += 8)
        xo[(size_t)(ty + yy) * CDIM + tx] = (__bf16)t[tx][ty + yy];
}

// ---------------------------------------------------------------------------
// Kernel 2: gather relative-position bias -> bias[h][n][m] f32 (2MB, L2-hot)
// ---------------------------------------------------------------------------
__global__ __launch_bounds__(256) void wa_bias(const float* __restrict__ table,
                                               const int* __restrict__ relidx,
                                               float* __restrict__ bias) {
    int nm = blockIdx.x * 256 + threadIdx.x;         // < 65536
    int idx = relidx[nm];
#pragma unroll
    for (int h = 0; h < NHEAD; ++h)
        bias[(size_t)h * 65536 + nm] = table[idx * NHEAD + h];
}

// ---------------------------------------------------------------------------
// Kernel 3: QKV GEMM.  y[b][o][hw] = sum_c Wb[o][c] * xT[b][hw][c], bf16 out.
// Grid (512, 12): per-block 128(M) x 256(N); 8 waves in 2x4 grid of 64x64.
// Fragments load straight from global (W 1.5MB + x[b] 256KB live in L2).
// ---------------------------------------------------------------------------
__global__ __launch_bounds__(256) void wa_gemm(const __bf16* __restrict__ Wb,
                                               const __bf16* __restrict__ xT,
                                               __bf16* __restrict__ y) {
    int b = blockIdx.x, mt = blockIdx.y;
    int tid = threadIdx.x, wave = tid >> 5, lane = tid & 31;
    int r = lane & 15, hf = lane >> 4;
    int m0 = mt * 128 + (wave >> 2) * 64;            // wave M base
    int n0 = (wave & 3) * 64;                        // wave N base
    const __bf16* xb = xT + (size_t)b * NTOK * CDIM;

    v8f acc[16];
#pragma unroll
    for (int i = 0; i < 16; ++i) acc[i] = (v8f){0.f,0.f,0.f,0.f,0.f,0.f,0.f,0.f};

    for (int k = 0; k < CDIM; k += 32) {
        v16bf af[4], bf[4];
#pragma unroll
        for (int i = 0; i < 4; ++i) {
            af[i] = ld_afrag(Wb + (size_t)(m0 + i * 16 + r) * CDIM + k, hf);
            bf[i] = ld_bfrag(xb + (size_t)(n0 + i * 16 + r) * CDIM + k, hf);
        }
#pragma unroll
        for (int i = 0; i < 4; ++i)
#pragma unroll
            for (int j = 0; j < 4; ++j)
                acc[i * 4 + j] = __builtin_amdgcn_wmma_f32_16x16x32_bf16(
                    false, af[i], false, bf[j], (short)0, acc[i * 4 + j],
                    false, false);
    }
    __bf16* yb = y + (size_t)b * TRIPLE * NTOK;
#pragma unroll
    for (int i = 0; i < 4; ++i)
#pragma unroll
        for (int j = 0; j < 4; ++j) {
            int mb = m0 + i * 16 + 8 * hf;           // C layout: row = e + 8*half
            int nn = n0 + j * 16 + r;                //           col = lane
#pragma unroll
            for (int e = 0; e < 8; ++e)
                yb[(size_t)(mb + e) * NTOK + nn] = (__bf16)acc[i * 4 + j][e];
        }
}

// ---------------------------------------------------------------------------
// Kernel 4: attention + BN.  One block of 16 waves per (b,h).
// Flat y reinterpreted: q/k/v rows at n*1536 + {0,512,1024} + h*64 (contig.)
// ---------------------------------------------------------------------------
__global__ __launch_bounds__(512) void wa_attn(const __bf16* __restrict__ y,
                                               const float* __restrict__ bias,
                                               const float* __restrict__ gamma,
                                               const float* __restrict__ beta,
                                               const float* __restrict__ mean,
                                               const float* __restrict__ var,
                                               float* __restrict__ out) {
    extern __shared__ char smem[];
    __bf16* VT  = (__bf16*)smem;                     // [64][VSTR]  V transposed
    __bf16* PST = (__bf16*)(smem + VT_BYTES);        // per-wave [16][PSTR]
    float*  OST = (float*)(smem + VT_BYTES + PST_BYTES); // [256][OSTR]

    int bh = blockIdx.x, b = bh >> 3, h = bh & 7;
    int tid = threadIdx.x, wave = tid >> 5, lane = tid & 31;
    int r = lane & 15, hf = lane >> 4;
    const __bf16* yb = y + (size_t)b * TRIPLE * NTOK + h * HDIM;

    // ---- stage V^T into LDS: VT[d][token] --------------------------------
    {
        int t = tid >> 1, dh = (tid & 1) * 32;
        const __bf16* pv = yb + (size_t)t * TRIPLE + 1024 + dh;
        V16 t0, t1;
        t0.u[0] = *(const uint4*)(pv);      t0.u[1] = *(const uint4*)(pv + 8);
        t1.u[0] = *(const uint4*)(pv + 16); t1.u[1] = *(const uint4*)(pv + 24);
#pragma unroll
        for (int e = 0; e < 16; ++e) {
            VT[(dh + e)      * VSTR + t] = t0.v[e];
            VT[(dh + 16 + e) * VSTR + t] = t1.v[e];
        }
    }
    __syncthreads();

    // ---- S = q * K^T  (wave owns 16 token rows) ---------------------------
    int n0 = wave * 16;
    v16bf qf[2];
#pragma unroll
    for (int ks = 0; ks < 2; ++ks)
        qf[ks] = ld_afrag(yb + (size_t)(n0 + r) * TRIPLE + ks * 32, hf);

    v8f s[16];
#pragma unroll
    for (int mi = 0; mi < 16; ++mi) {
        v8f a = (v8f){0.f,0.f,0.f,0.f,0.f,0.f,0.f,0.f};
#pragma unroll
        for (int ks = 0; ks < 2; ++ks) {
            v16bf kf = ld_bfrag(yb + (size_t)(mi * 16 + r) * TRIPLE + 512 + ks * 32, hf);
            a = __builtin_amdgcn_wmma_f32_16x16x32_bf16(false, qf[ks], false, kf,
                                                        (short)0, a, false, false);
        }
        s[mi] = a;
    }

    // ---- + relative-position bias ----------------------------------------
    const float* bh_bias = bias + (size_t)h * 65536;
#pragma unroll
    for (int mi = 0; mi < 16; ++mi)
#pragma unroll
        for (int e = 0; e < 8; ++e)
            s[mi][e] += bh_bias[(size_t)(n0 + e + 8 * hf) * NTOK + mi * 16 + r];

    // ---- softmax: row = (e,half); reduce over mi (regs) + 16 lanes -------
    float inv[8];
#pragma unroll
    for (int e = 0; e < 8; ++e) {
        float mx = s[0][e];
#pragma unroll
        for (int mi = 1; mi < 16; ++mi) mx = fmaxf(mx, s[mi][e]);
#pragma unroll
        for (int msk = 1; msk <= 8; msk <<= 1)
            mx = fmaxf(mx, __shfl_xor(mx, msk, 32));
        float sm = 0.f;
#pragma unroll
        for (int mi = 0; mi < 16; ++mi) { s[mi][e] = __expf(s[mi][e] - mx); sm += s[mi][e]; }
#pragma unroll
        for (int msk = 1; msk <= 8; msk <<= 1)
            sm += __shfl_xor(sm, msk, 32);
        inv[e] = 1.0f / sm;
    }

    // ---- O = P * V  (P through per-wave LDS: C-layout -> A-layout) -------
    __bf16* pw = PST + (size_t)wave * 16 * PSTR;
    v8f o[4];
#pragma unroll
    for (int dt = 0; dt < 4; ++dt) o[dt] = (v8f){0.f,0.f,0.f,0.f,0.f,0.f,0.f,0.f};

    for (int ch = 0; ch < 8; ++ch) {                 // 32 m-columns per chunk
#pragma unroll
        for (int sub = 0; sub < 2; ++sub) {
            int mi = ch * 2 + sub;
#pragma unroll
            for (int e = 0; e < 8; ++e)
                pw[(e + 8 * hf) * PSTR + sub * 16 + r] =
                    (__bf16)(s[mi][e] * inv[e]);
        }
        v16bf pf = ld_afrag(pw + r * PSTR, hf);      // compiler inserts s_wait_dscnt
#pragma unroll
        for (int dt = 0; dt < 4; ++dt) {
            v16bf vf = ld_bfrag(VT + (size_t)(dt * 16 + r) * VSTR + ch * 32, hf);
            o[dt] = __builtin_amdgcn_wmma_f32_16x16x32_bf16(false, pf, false, vf,
                                                            (short)0, o[dt],
                                                            false, false);
        }
    }

    // ---- stage O to LDS, then coalesced BN + store -----------------------
#pragma unroll
    for (int dt = 0; dt < 4; ++dt)
#pragma unroll
        for (int e = 0; e < 8; ++e)
            OST[(size_t)(n0 + e + 8 * hf) * OSTR + dt * 16 + r] = o[dt][e];
    __syncthreads();

#pragma unroll
    for (int g = 0; g < 8; ++g) {
        int G = tid + 512 * g;                       // < 4096
        int d = G >> 6, nb = (G & 63) * 4;
        int c = h * HDIM + d;
        float iv = gamma[c] * rsqrtf(var[c] + 1e-5f);
        float sh = beta[c] - mean[c] * iv;
        float4 vv;
        vv.x = OST[(size_t)(nb + 0) * OSTR + d] * iv + sh;
        vv.y = OST[(size_t)(nb + 1) * OSTR + d] * iv + sh;
        vv.z = OST[(size_t)(nb + 2) * OSTR + d] * iv + sh;
        vv.w = OST[(size_t)(nb + 3) * OSTR + d] * iv + sh;
        *(float4*)(out + ((size_t)b * CDIM + c) * NTOK + nb) = vv;
    }
}

// ---------------------------------------------------------------------------
extern "C" void kernel_launch(void* const* d_in, const int* in_sizes, int n_in,
                              void* d_out, int out_size, void* d_ws, size_t ws_size,
                              hipStream_t stream) {
    (void)in_sizes; (void)n_in; (void)out_size; (void)ws_size;
    const float* x      = (const float*)d_in[0];
    const float* w_qkv  = (const float*)d_in[1];
    const float* table  = (const float*)d_in[2];
    const int*   relidx = (const int*)d_in[3];
    const float* gamma  = (const float*)d_in[4];
    const float* beta   = (const float*)d_in[5];
    const float* mean   = (const float*)d_in[6];
    const float* var    = (const float*)d_in[7];
    float* out = (float*)d_out;

    char* ws = (char*)d_ws;
    __bf16* Wb   = (__bf16*)(ws);
    __bf16* xT   = (__bf16*)(ws + WB_BYTES);
    float*  bias = (float*) (ws + WB_BYTES + XT_BYTES);
    __bf16* y    = (__bf16*)(ws + WB_BYTES + XT_BYTES + BIAS_BYTES);

    wa_cvt_w <<<(TRIPLE * CDIM) / 256, 256, 0, stream>>>(w_qkv, Wb);
    wa_cvt_xT<<<dim3(NTOK / 32, CDIM / 32, BATCH), dim3(32, 8), 0, stream>>>(x, xT);
    wa_bias  <<<(NTOK * NTOK) / 256, 256, 0, stream>>>(table, relidx, bias);
    wa_gemm  <<<dim3(BATCH, TRIPLE / 128), 256, 0, stream>>>(Wb, xT, y);
    wa_attn  <<<BATCH * NHEAD, 512, ATTN_SMEM, stream>>>(y, bias, gamma, beta,
                                                         mean, var, out);
}
